// Hawk_83777632076219
// MI455X (gfx1250) — matrix operand order repaired and checked by reading, not compiled, not run on hardware
//
#include <hip/hip_runtime.h>
#include <math.h>

// ---------------------------------------------------------------------------
// Hawk / RG-LRU single step for MI455X (gfx1250, wave32, WMMA bf16).
//   B=4096, H=2048, K_taps=4, NB=4, Hb=512
// GEMMs: v_wmma_f32_16x16x32_bf16, 256x128 WG tile, 64x64 wave tile,
// double-buffered LDS fed by global_load_async_to_lds_b128 (ASYNCcnt).
// ---------------------------------------------------------------------------

typedef __attribute__((ext_vector_type(16))) __bf16 v16bf;
typedef __attribute__((ext_vector_type(8)))  __bf16 v8bf;
typedef __attribute__((ext_vector_type(4)))  __bf16 v4bf;
typedef __attribute__((ext_vector_type(8)))  float  v8f;
typedef __attribute__((ext_vector_type(4)))  int    v4i;

// builtin signature (from hipcc diagnostic): (as1 v4i*, as3 v4i*, imm, imm)
typedef __attribute__((address_space(1))) v4i glob_v4i;
typedef __attribute__((address_space(3))) v4i lds_v4i;

static constexpr int BQ  = 4096;   // batch
static constexpr int H   = 2048;   // hidden
static constexpr int HB  = 512;    // block hidden
static constexpr int NB  = 4;      // blocks

// --- CDNA5 async copy: global -> LDS, 16 bytes, tracked by ASYNCcnt --------
__device__ __forceinline__ void async_copy_b128(const __bf16* g, __bf16* l) {
#if __has_builtin(__builtin_amdgcn_global_load_async_to_lds_b128)
  __builtin_amdgcn_global_load_async_to_lds_b128(
      (glob_v4i*)(unsigned long long)(uintptr_t)g,
      (lds_v4i*)(unsigned)(uintptr_t)l, 0, 0);
#else
  *(uint4*)l = *(const uint4*)g;     // synchronous fallback, same control flow
#endif
}

__device__ __forceinline__ void wait_async0() {
#if __has_builtin(__builtin_amdgcn_s_wait_asynccnt)
  __builtin_amdgcn_s_wait_asynccnt(0);
#else
  asm volatile("s_wait_asynccnt 0x0" ::: "memory");
#endif
}

// ---------------------------------------------------------------------------
// f32 -> bf16 conversion, 4 elements / thread
// ---------------------------------------------------------------------------
__global__ __launch_bounds__(256) void cvt_f32_bf16_kernel(
    const float* __restrict__ in, __bf16* __restrict__ out, int n4) {
  int i = blockIdx.x * 256 + threadIdx.x;
  if (i < n4) {
    float4 v = ((const float4*)in)[i];
    v4bf o;
    o[0] = (__bf16)v.x; o[1] = (__bf16)v.y;
    o[2] = (__bf16)v.z; o[3] = (__bf16)v.w;
    ((v4bf*)out)[i] = o;
  }
}

__device__ __forceinline__ float gelu_exact(float x) {
  return 0.5f * x * (1.0f + erff(x * 0.70710678118654752f));
}

// ---------------------------------------------------------------------------
// bf16 WMMA GEMM:  C[m,n] = sum_k A[m,k] * B[n,k]   (B row-major = W)
// 256 thr = 8 waves (4M x 2N); WG tile 256(M) x 128(N); wave tile 64x64.
// Double-buffered LDS, async global->LDS tile loads.
// blockIdx.z selects a diagonal block (zA/zB/zC element offsets per z).
// epi == 0 : C0[z*zC + m*ldc + n] = acc
// epi == 1 : n <  H : C0[m*H + n]             = gelu(acc)   (gate)
//            n >= H : C1[m*3H + 2H + (n - H)] = acc         (xr -> d_out)
// ---------------------------------------------------------------------------
__global__ __launch_bounds__(256) void gemm_bf16_wmma_kernel(
    const __bf16* __restrict__ A, const __bf16* __restrict__ B,
    float* __restrict__ C0, float* __restrict__ C1,
    int Kd, int lda, int ldb, int ldc,
    int zA, int zB, int zC, int epi) {
  constexpr int BM = 256, BN = 128;
  constexpr int LSTR = 40;                 // padded LDS row stride (bf16)
  __shared__ __bf16 sA[2][BM * LSTR];      // 2 x 20 KB
  __shared__ __bf16 sB[2][BN * LSTR];      // 2 x 10 KB

  const int tid  = threadIdx.x;
  const int lane = tid & 31;
  const int w    = tid >> 5;
  const int wm   = w >> 1;                 // 0..3 -> M offset wm*64
  const int wn   = w & 1;                  // 0..1 -> N offset wn*64
  const int l15  = lane & 15;
  const int lhi  = lane >> 4;              // 0..1

  const int blockM = blockIdx.x * BM;
  const int blockN = blockIdx.y * BN;
  const int z      = blockIdx.z;

  const __bf16* Ab = A + (size_t)z * zA;
  const __bf16* Bb = B + (size_t)z * zB;

  v8f zero = {};
  v8f acc[4][4];
  for (int mi = 0; mi < 4; ++mi)
    for (int ni = 0; ni < 4; ++ni) acc[mi][ni] = zero;

  // per-thread tile-load assignment (16B chunks):
  //   A: row = tid (256 rows), 4 chunks of its 32-elem row
  //   B: row = tid>>1 (128 rows), 2 chunks at (tid&1)*16
  const int brow = tid >> 1;
  const int bcol = (tid & 1) * 16;

  auto issue_tile = [&](int kt, int buf) {
    const __bf16* aSrc = Ab + (size_t)(blockM + tid) * lda + kt;
    __bf16* aDst = &sA[buf][tid * LSTR];
#pragma unroll
    for (int j = 0; j < 4; ++j) async_copy_b128(aSrc + j * 8, aDst + j * 8);
    const __bf16* bSrc = Bb + (size_t)(blockN + brow) * ldb + kt + bcol;
    __bf16* bDst = &sB[buf][brow * LSTR + bcol];
    async_copy_b128(bSrc, bDst);
    async_copy_b128(bSrc + 8, bDst + 8);
  };

  issue_tile(0, 0);

  for (int kt = 0, it = 0; kt < Kd; kt += 32, ++it) {
    const int cur = it & 1;
    wait_async0();                 // tile `it` fully landed in LDS
    __syncthreads();               // ... for every wave in the group

    // Fragment layouts per CDNA5 ISA 7.12.2 (16-bit, 16x16x32):
    //  A: lane l -> row l&15; 8 contiguous K at (l>>4)*8 and +16
    //  B: lane l -> col l&15; 8 contiguous K at (l>>4)*8 and +16
    union { v16bf v; v8bf h[2]; } af[4], bfr[4];
#pragma unroll
    for (int mi = 0; mi < 4; ++mi) {
      int r = wm * 64 + mi * 16 + l15;
      af[mi].h[0] = *(const v8bf*)&sA[cur][r * LSTR + lhi * 8];
      af[mi].h[1] = *(const v8bf*)&sA[cur][r * LSTR + lhi * 8 + 16];
    }
#pragma unroll
    for (int ni = 0; ni < 4; ++ni) {
      int n = wn * 64 + ni * 16 + l15;
      bfr[ni].h[0] = *(const v8bf*)&sB[cur][n * LSTR + lhi * 8];
      bfr[ni].h[1] = *(const v8bf*)&sB[cur][n * LSTR + lhi * 8 + 16];
    }

    if (kt + 32 < Kd) issue_tile(kt + 32, cur ^ 1);  // prefetch behind WMMA

#pragma unroll
    for (int mi = 0; mi < 4; ++mi)
#pragma unroll
      for (int ni = 0; ni < 4; ++ni)
        acc[mi][ni] = __builtin_amdgcn_wmma_f32_16x16x32_bf16(
            false, af[mi].v, false, bfr[ni].v,
            (short)0, acc[mi][ni], false, false);
  }

  // Epilogue. C layout: lane l holds n = l&15, rows m = (l>>4)*8 + r.
#pragma unroll
  for (int mi = 0; mi < 4; ++mi) {
#pragma unroll
    for (int ni = 0; ni < 4; ++ni) {
      int mBase = blockM + wm * 64 + mi * 16 + lhi * 8;
      int nIdx  = blockN + wn * 64 + ni * 16 + l15;
      if (epi == 0) {
        float* Cp = C0 + (size_t)z * zC + nIdx;
        for (int r = 0; r < 8; ++r)
          Cp[(size_t)(mBase + r) * ldc] = acc[mi][ni][r];
      } else {
        if (nIdx < H) {
          for (int r = 0; r < 8; ++r)
            C0[(size_t)(mBase + r) * H + nIdx] = gelu_exact(acc[mi][ni][r]);
        } else {
          int hh = nIdx - H;
          for (int r = 0; r < 8; ++r)   // xr -> new_conv_state[:, 2, :]
            C1[(size_t)(mBase + r) * (3 * H) + 2 * H + hh] = acc[mi][ni][r];
        }
      }
    }
  }
}

// ---------------------------------------------------------------------------
// Conv over 4 taps + conv-state shift.  xr already sits in convOut[:,2,:].
// ---------------------------------------------------------------------------
__global__ __launch_bounds__(256) void conv_kernel(
    const float* __restrict__ cs, const float* __restrict__ cw,
    const float* __restrict__ cb, float* __restrict__ convOut,
    float* __restrict__ xc, __bf16* __restrict__ xcbf) {
  int idx = blockIdx.x * 256 + threadIdx.x;      // [0, B*H)
  int b = idx >> 11;
  int h = idx & (H - 1);
  size_t base = (size_t)b * (3 * H);
  float c0 = cs[base + h];
  float c1 = cs[base + H + h];
  float c2 = cs[base + 2 * H + h];
  float xr = convOut[base + 2 * H + h];
  float v = c0 * cw[h] + c1 * cw[H + h] + c2 * cw[2 * H + h]
          + xr * cw[3 * H + h] + cb[h];
  convOut[base + h]     = c1;                    // shifted state
  convOut[base + H + h] = c2;
  xc[idx]   = v;
  xcbf[idx] = (__bf16)v;
}

// ---------------------------------------------------------------------------
// RG-LRU elementwise:  a_t = a^(8*sigmoid(v));  state' = s*a_t + m*i_t*xc
// Also forms (gate * state') in bf16 for the output GEMM.
// ---------------------------------------------------------------------------
__global__ __launch_bounds__(256) void rglru_kernel(
    const float* __restrict__ u, const float* __restrict__ v,
    const float* __restrict__ xc, const float* __restrict__ sin_,
    const float* __restrict__ a, const float* __restrict__ gate,
    float* __restrict__ sout, __bf16* __restrict__ gbf) {
  int idx = blockIdx.x * 256 + threadIdx.x;
  int h = idx & (H - 1);
  float it = 1.0f / (1.0f + __expf(-u[idx]));
  float rt = 1.0f / (1.0f + __expf(-v[idx]));
  float av = a[h];                               // in (0.5, 0.99): log safe
  float at = __expf(8.0f * rt * __logf(av));
  float mult = sqrtf(fmaxf(0.0f, 1.0f - at * at));
  float ns = sin_[idx] * at + mult * (it * xc[idx]);
  sout[idx] = ns;
  gbf[idx]  = (__bf16)(gate[idx] * ns);
}

// ---------------------------------------------------------------------------
extern "C" void kernel_launch(void* const* d_in, const int* in_sizes, int n_in,
                              void* d_out, int out_size, void* d_ws, size_t ws_size,
                              hipStream_t stream) {
  const float* x   = (const float*)d_in[0];
  const float* cs  = (const float*)d_in[1];
  const float* st  = (const float*)d_in[2];
  const float* Wg  = (const float*)d_in[3];
  const float* Wr  = (const float*)d_in[4];
  const float* Wo  = (const float*)d_in[5];
  const float* cw  = (const float*)d_in[6];
  const float* cb  = (const float*)d_in[7];
  const float* Wi  = (const float*)d_in[8];
  const float* Wrg = (const float*)d_in[9];
  const float* a   = (const float*)d_in[10];

  float* out   = (float*)d_out;                     // [B,H]
  float* conv  = out + (size_t)BQ * H;              // [B,3,H]
  float* sout  = conv + (size_t)BQ * 3 * H;         // [B,H]

  // ---- workspace layout (bytes); total ~173 MB ----
  char* ws = (char*)d_ws;
  const size_t SZ_XBF  = (size_t)BQ * H * 2;        // 16 MiB
  const size_t SZ_WGR  = (size_t)2 * H * H * 2;     // 16 MiB  (Wg ; Wr)
  const size_t SZ_WO   = (size_t)H * H * 2;         //  8 MiB
  const size_t SZ_WI   = (size_t)NB * HB * HB * 2;  //  2 MiB
  const size_t SZ_F32  = (size_t)BQ * H * 4;        // 32 MiB

  __bf16* xbf   = (__bf16*)(ws);
  __bf16* wgr   = (__bf16*)(ws + SZ_XBF);
  __bf16* wobf  = (__bf16*)(ws + SZ_XBF + SZ_WGR);
  __bf16* wibf  = (__bf16*)(ws + SZ_XBF + SZ_WGR + SZ_WO);
  __bf16* wrgbf = (__bf16*)(ws + SZ_XBF + SZ_WGR + SZ_WO + SZ_WI);
  size_t  o0    = SZ_XBF + SZ_WGR + SZ_WO + 2 * SZ_WI;
  float*  gate  = (float*)(ws + o0);
  float*  xc    = (float*)(ws + o0 + SZ_F32);
  __bf16* xcbf  = (__bf16*)(ws + o0 + 2 * SZ_F32);
  float*  vbuf  = (float*)(ws + o0 + 2 * SZ_F32 + SZ_XBF);
  __bf16* gbf   = (__bf16*)(ws + o0 + 3 * SZ_F32 + SZ_XBF);
  float*  ubuf  = (float*)(ws);                     // alias: xbf+wgr dead after GEMM1

  const dim3 tb(256);

  // 0) f32 -> bf16 conversions
  cvt_f32_bf16_kernel<<<dim3((BQ * H) / 1024), tb, 0, stream>>>(x, xbf, (BQ * H) / 4);
  cvt_f32_bf16_kernel<<<dim3((H * H) / 1024), tb, 0, stream>>>(Wg, wgr, (H * H) / 4);
  cvt_f32_bf16_kernel<<<dim3((H * H) / 1024), tb, 0, stream>>>(Wr, wgr + (size_t)H * H, (H * H) / 4);
  cvt_f32_bf16_kernel<<<dim3((H * H) / 1024), tb, 0, stream>>>(Wo, wobf, (H * H) / 4);
  cvt_f32_bf16_kernel<<<dim3((NB * HB * HB) / 1024), tb, 0, stream>>>(Wi, wibf, (NB * HB * HB) / 4);
  cvt_f32_bf16_kernel<<<dim3((NB * HB * HB) / 1024), tb, 0, stream>>>(Wrg, wrgbf, (NB * HB * HB) / 4);

  // 1) fused GEMM: x @ [Wg;Wr]^T -> gelu->gate, xr->conv[:,2,:]
  gemm_bf16_wmma_kernel<<<dim3(BQ / 256, (2 * H) / 128, 1), tb, 0, stream>>>(
      xbf, wgr, gate, conv, H, H, H, H, 0, 0, 0, /*epi=*/1);

  // 2) conv taps + state shift
  conv_kernel<<<dim3((BQ * H) / 256), tb, 0, stream>>>(cs, cw, cb, conv, xc, xcbf);

  // 3) block-diagonal GEMMs: u = xc @ Wi^T, v = xc @ Wrg^T  (grid.z = block)
  gemm_bf16_wmma_kernel<<<dim3(BQ / 256, HB / 128, NB), tb, 0, stream>>>(
      xcbf, wibf, ubuf, nullptr, HB, H, HB, H, HB, HB * HB, HB, 0);
  gemm_bf16_wmma_kernel<<<dim3(BQ / 256, HB / 128, NB), tb, 0, stream>>>(
      xcbf, wrgbf, vbuf, nullptr, HB, H, HB, H, HB, HB * HB, HB, 0);

  // 4) RG-LRU elementwise, emit new state + (gate*state) bf16
  rglru_kernel<<<dim3((BQ * H) / 256), tb, 0, stream>>>(
      ubuf, vbuf, xc, st, a, gate, sout, gbf);

  // 5) output GEMM: (gate*state) @ Wo^T -> out
  gemm_bf16_wmma_kernel<<<dim3(BQ / 256, H / 128, 1), tb, 0, stream>>>(
      gbf, wobf, out, nullptr, H, H, H, H, 0, 0, 0, 0);
}
